// positional_spiking_attention_27462020890903
// MI455X (gfx1250) — matrix operand order repaired and checked by reading, not compiled, not run on hardware
//
#include <hip/hip_runtime.h>

// ---------------------------------------------------------------------------
// Spiking positional attention, fused for MI455X (gfx1250, wave32, WMMA).
//   T=4 B=4 L=1024 D=512 H=8 WINDOW=8
//   Pipeline: lif(x) -> 3x [f16 WMMA GEMM + BN] -> lif -> banded conv -> lif
//             -> final f16 WMMA GEMM + BN -> d_out (fp32)
// GEMM staging uses gfx1250 async global->LDS DMA with double buffering.
// All GEMM activations are binary spikes -> exact in f16.
// ---------------------------------------------------------------------------

typedef __attribute__((ext_vector_type(16))) _Float16 v16h;
typedef __attribute__((ext_vector_type(8)))  float    v8f;

#define T_DIM 4
#define B_DIM 4
#define L_DIM 1024
#define D_DIM 512
#define M_DIM (T_DIM * B_DIM * L_DIM)     // 16384 GEMM rows
#define BLD   (B_DIM * L_DIM * D_DIM)     // 2097152 elems per time step
#define INV_STD 0.9999950000374997f       // 1/sqrt(1+1e-5)

// async global->LDS copy: INST_OFFSET is added to BOTH lds and global address
#define ASYNC_CP_B128(ldsoff, gptr)                                      \
  asm volatile("global_load_async_to_lds_b128 %0, %1, off"               \
               :: "v"(ldsoff), "v"(gptr) : "memory")
#define ASYNC_CP_B128_O16(ldsoff, gptr)                                  \
  asm volatile("global_load_async_to_lds_b128 %0, %1, off offset:16"     \
               :: "v"(ldsoff), "v"(gptr) : "memory")
#define WAIT_ASYNC0() asm volatile("s_wait_asynccnt 0" ::: "memory")

// ---------------------------- elementwise kernels ---------------------------

// weights fp32 -> f16, 4 elements / thread (b128 in, b64 out)
__global__ void f32_to_f16_kernel(const float* __restrict__ in,
                                  _Float16* __restrict__ out) {
  size_t i = ((size_t)blockIdx.x * blockDim.x + threadIdx.x) * 4;
  float4 v = *(const float4*)(in + i);
  union { _Float16 h[4]; uint2 u; } o;
  o.h[0] = (_Float16)v.x; o.h[1] = (_Float16)v.y;
  o.h[2] = (_Float16)v.z; o.h[3] = (_Float16)v.w;
  *(uint2*)(out + i) = o.u;
}

// LIF over T=4: mem_t = mem_{t-1}*0.25*(1-spike_{t-1}) + x_t ; spike = mem_t>0.5
// in: fp32 [T, BLD] (t-major), out: f16 spikes same layout. 4 elems/thread.
__global__ void lif_kernel(const float* __restrict__ in,
                           _Float16* __restrict__ spikes) {
  size_t i = ((size_t)blockIdx.x * blockDim.x + threadIdx.x) * 4;
  float mem[4], sp[4];
  union { _Float16 h[4]; uint2 u; } o;
  float4 v = *(const float4*)(in + i);
  mem[0] = v.x; mem[1] = v.y; mem[2] = v.z; mem[3] = v.w;
#pragma unroll
  for (int j = 0; j < 4; ++j) {
    sp[j] = mem[j] > 0.5f ? 1.0f : 0.0f;
    o.h[j] = (_Float16)sp[j];
  }
  *(uint2*)(spikes + i) = o.u;
#pragma unroll
  for (int t = 1; t < T_DIM; ++t) {
    v = *(const float4*)(in + (size_t)t * BLD + i);
    float xt[4] = {v.x, v.y, v.z, v.w};
#pragma unroll
    for (int j = 0; j < 4; ++j) {
      mem[j] = mem[j] * 0.25f * (1.0f - sp[j]) + xt[j];
      sp[j] = mem[j] > 0.5f ? 1.0f : 0.0f;
      o.h[j] = (_Float16)sp[j];
    }
    *(uint2*)(spikes + (size_t)t * BLD + i) = o.u;
  }
}

// Banded positional mixing (4 elems/thread, never crosses an l boundary):
//   out[t,b,l,d] = q[t,b,l,d] * sum_{w=0..7, l-w>=0} pb[l,l-w]*k[..l-w..]*v[..l-w..]
__global__ void attn_band_kernel(const _Float16* __restrict__ q_s,
                                 const _Float16* __restrict__ k_s,
                                 const _Float16* __restrict__ v_s,
                                 const float* __restrict__ pb,
                                 float* __restrict__ out) {
  size_t i = ((size_t)blockIdx.x * blockDim.x + threadIdx.x) * 4;
  int l = (int)((i >> 9) & (L_DIM - 1));
  int wmax = l < 7 ? l : 7;          // uniform across a wave (128 d per wave)
  float sum[4] = {0.f, 0.f, 0.f, 0.f};
  for (int w = 0; w <= wmax; ++w) {
    size_t s = i - (size_t)w * D_DIM;
    union { uint2 u; _Float16 h[4]; } kk, vv;
    kk.u = *(const uint2*)(k_s + s);
    vv.u = *(const uint2*)(v_s + s);
    float p = pb[(size_t)l * L_DIM + (l - w)];
#pragma unroll
    for (int j = 0; j < 4; ++j)
      sum[j] += p * (float)kk.h[j] * (float)vv.h[j];
  }
  union { uint2 u; _Float16 h[4]; } qq;
  qq.u = *(const uint2*)(q_s + i);
  float4 o;
  o.x = (float)qq.h[0] * sum[0];
  o.y = (float)qq.h[1] * sum[1];
  o.z = (float)qq.h[2] * sum[2];
  o.w = (float)qq.h[3] * sum[3];
  *(float4*)(out + i) = o;
}

// ------------------------------- WMMA GEMM ----------------------------------
// C[M=16384, N=512] = A[M,512] (f16 spikes) @ W[N,512]^T  with fused
// (acc + bias) * (INV_STD*gamma) + beta epilogue, fp32 output.
// Block: 256 threads (8 waves), 128x128 tile; wave = 32x64 (2x4 WMMA frags).
// Staging: double-buffered async global->LDS DMA (ASYNCcnt).

#define BM 128
#define BN 128
#define BK 32
#define LDPAD 40                       // halves per LDS row (16B-aligned rows)
#define TILE_HALVES (BM * LDPAD)       // 5120 halves = 10240 bytes per buffer

__global__ __launch_bounds__(256)
void gemm_bn_kernel(const _Float16* __restrict__ A,
                    const _Float16* __restrict__ W,   // [N=512][K=512] row-major
                    const float* __restrict__ bias,
                    const float* __restrict__ gamma,
                    const float* __restrict__ beta,
                    float* __restrict__ out) {
  __shared__ __align__(16) _Float16 ldsA[2 * TILE_HALVES];  // A tiles (2 bufs)
  __shared__ __align__(16) _Float16 ldsB[2 * TILE_HALVES];  // W tiles (2 bufs)

  const int tid  = threadIdx.x;
  const int lane = tid & 31;
  const int wid  = tid >> 5;
  const int wm   = wid & 3;     // wave M band (4 bands of 32 rows)
  const int wn   = wid >> 2;    // wave N band (2 bands of 64 cols)
  const int m16  = lane & 15;
  const int kh   = lane >> 4;   // lane half-group
  const int blockM = blockIdx.y * BM;
  const int blockN = blockIdx.x * BN;

  // staging: both tiles are 128 rows x 32 halves; thread: row=tid>>1, 16 halves
  const int srow = tid >> 1, shc = tid & 1;
  const _Float16* gA = A + (size_t)(blockM + srow) * D_DIM + shc * 16;
  const _Float16* gW = W + (size_t)(blockN + srow) * D_DIM + shc * 16;
  // LDS byte offset = low 32 bits of the flat shared-memory address
  const unsigned laOff = (unsigned)(size_t)(ldsA + srow * LDPAD + shc * 16);
  const unsigned lbOff = (unsigned)(size_t)(ldsB + srow * LDPAD + shc * 16);
  const unsigned bufBytes = TILE_HALVES * 2;   // 10240

  union Frag { v16h v; uint4 u[2]; };
  v8f acc[2][4] = {};

  // prologue: async-stage K-tile 0 into buffer 0
  {
    ASYNC_CP_B128(laOff, gA);
    ASYNC_CP_B128_O16(laOff, gA);
    ASYNC_CP_B128(lbOff, gW);
    ASYNC_CP_B128_O16(lbOff, gW);
  }

  int cur = 0;
  for (int k0 = 0; k0 < D_DIM; k0 += BK) {
    WAIT_ASYNC0();          // my DMA slices for buf[cur] have landed
    __syncthreads();        // everyone's slices have landed

    if (k0 + BK < D_DIM) {  // async-stage next K-tile into the other buffer
      const _Float16* ga = gA + k0 + BK;
      const _Float16* gw = gW + k0 + BK;
      unsigned la = laOff + (unsigned)(cur ^ 1) * bufBytes;
      unsigned lb = lbOff + (unsigned)(cur ^ 1) * bufBytes;
      ASYNC_CP_B128(la, ga);
      ASYNC_CP_B128_O16(la, ga);
      ASYNC_CP_B128(lb, gw);
      ASYNC_CP_B128_O16(lb, gw);
    }
    if (k0 + 2 * BK < D_DIM) {  // warm L2 two tiles ahead (global_prefetch_b8)
      __builtin_prefetch(gA + k0 + 2 * BK, 0, 1);
      __builtin_prefetch(gW + k0 + 2 * BK, 0, 1);
    }

    const _Float16* bA = ldsA + cur * TILE_HALVES;
    const _Float16* bB = ldsB + cur * TILE_HALVES;

    // ---- gather fragments per ISA VGPR layouts (wave32)
    Frag a[2], b[4];
#pragma unroll
    for (int f = 0; f < 2; ++f) {
      // A 16x32 f16: lanes 0-15 row M=lane K0-7/16-23; lanes 16-31 K8-15/24-31
      const _Float16* ap = bA + (wm * 32 + f * 16 + m16) * LDPAD + kh * 8;
      a[f].u[0] = *(const uint4*)ap;          // K = kh*8 .. +7
      a[f].u[1] = *(const uint4*)(ap + 16);   // K = 16 + kh*8 .. +7
    }
#pragma unroll
    for (int f = 0; f < 4; ++f) {
      // B 32x16 f16 (stored transposed): lane n=lane&15, K = kh*8 .. +7, 16 + kh*8 .. +7
      const _Float16* bp = bB + (wn * 64 + f * 16 + m16) * LDPAD + kh * 8;
      b[f].u[0] = *(const uint4*)bp;
      b[f].u[1] = *(const uint4*)(bp + 16);
    }

#pragma unroll
    for (int fm = 0; fm < 2; ++fm)
#pragma unroll
      for (int fn = 0; fn < 4; ++fn)
        acc[fm][fn] = __builtin_amdgcn_wmma_f32_16x16x32_f16(
            false, a[fm].v, false, b[fn].v, (short)0, acc[fm][fn],
            false, false);
    cur ^= 1;
  }

  // ---- fused bias + BN(eval) epilogue; C layout: lane n=lane&15, M=r+8*kh
#pragma unroll
  for (int fm = 0; fm < 2; ++fm) {
#pragma unroll
    for (int fn = 0; fn < 4; ++fn) {
      int ng = blockN + wn * 64 + fn * 16 + m16;
      float sc = INV_STD * gamma[ng];
      float bi = bias[ng];
      float be = beta[ng];
      int mg = blockM + wm * 32 + fm * 16 + kh * 8;
#pragma unroll
      for (int r = 0; r < 8; ++r) {
        float v = (acc[fm][fn][r] + bi) * sc + be;
        out[(size_t)(mg + r) * D_DIM + ng] = v;
      }
    }
  }
}

// ------------------------------ launch ----------------------------------

extern "C" void kernel_launch(void* const* d_in, const int* in_sizes, int n_in,
                              void* d_out, int out_size, void* d_ws,
                              size_t ws_size, hipStream_t stream) {
  (void)in_sizes; (void)n_in; (void)out_size; (void)ws_size;

  const float* x        = (const float*)d_in[0];
  const float* pos_bias = (const float*)d_in[1];
  const float* q_w  = (const float*)d_in[2];
  const float* q_b  = (const float*)d_in[3];
  const float* q_g  = (const float*)d_in[4];
  const float* q_be = (const float*)d_in[5];
  const float* k_w  = (const float*)d_in[6];
  const float* k_b  = (const float*)d_in[7];
  const float* k_g  = (const float*)d_in[8];
  const float* k_be = (const float*)d_in[9];
  const float* v_w  = (const float*)d_in[10];
  const float* v_b  = (const float*)d_in[11];
  const float* v_g  = (const float*)d_in[12];
  const float* v_be = (const float*)d_in[13];
  const float* l_w  = (const float*)d_in[14];
  const float* l_b  = (const float*)d_in[15];
  const float* l_g  = (const float*)d_in[16];
  const float* l_be = (const float*)d_in[17];

  // workspace layout (bytes)
  char* ws = (char*)d_ws;
  _Float16* wq16   = (_Float16*)(ws + 0);          // 512KB each
  _Float16* wk16   = (_Float16*)(ws + 524288);
  _Float16* wv16   = (_Float16*)(ws + 1048576);
  _Float16* wl16   = (_Float16*)(ws + 1572864);
  _Float16* xs     = (_Float16*)(ws + 2097152);    // 16MB spikes (reused as s2)
  _Float16* q_s    = (_Float16*)(ws + 18874368);   // 16MB
  _Float16* k_s    = (_Float16*)(ws + 35651584);   // 16MB
  _Float16* v_s    = (_Float16*)(ws + 52428800);   // 16MB
  float*    preact = (float*)   (ws + 69206016);   // 32MB fp32 (reused)
  float*    out    = (float*)d_out;

  const int WCVT_BLOCKS = (D_DIM * D_DIM) / (256 * 4);   // 256
  const int LIF_BLOCKS  = BLD / (256 * 4);               // 2048
  const int EW_BLOCKS   = (T_DIM * BLD) / (256 * 4);     // 8192
  dim3 ggrid(D_DIM / BN, M_DIM / BM);                    // (4, 128)

  // weights -> f16 once per launch
  f32_to_f16_kernel<<<WCVT_BLOCKS, 256, 0, stream>>>(q_w, wq16);
  f32_to_f16_kernel<<<WCVT_BLOCKS, 256, 0, stream>>>(k_w, wk16);
  f32_to_f16_kernel<<<WCVT_BLOCKS, 256, 0, stream>>>(v_w, wv16);
  f32_to_f16_kernel<<<WCVT_BLOCKS, 256, 0, stream>>>(l_w, wl16);

  // first LIF: x -> spikes xs (f16, binary)
  lif_kernel<<<LIF_BLOCKS, 256, 0, stream>>>(x, xs);

  // q / k / v projections: WMMA GEMM + fused BN, then LIF -> spikes
  gemm_bn_kernel<<<ggrid, 256, 0, stream>>>(xs, wq16, q_b, q_g, q_be, preact);
  lif_kernel<<<LIF_BLOCKS, 256, 0, stream>>>(preact, q_s);
  gemm_bn_kernel<<<ggrid, 256, 0, stream>>>(xs, wk16, k_b, k_g, k_be, preact);
  lif_kernel<<<LIF_BLOCKS, 256, 0, stream>>>(preact, k_s);
  gemm_bn_kernel<<<ggrid, 256, 0, stream>>>(xs, wv16, v_b, v_g, v_be, preact);
  lif_kernel<<<LIF_BLOCKS, 256, 0, stream>>>(preact, v_s);

  // banded positional mixing + q gating (fp32, into preact)
  attn_band_kernel<<<EW_BLOCKS, 256, 0, stream>>>(q_s, k_s, v_s, pos_bias,
                                                  preact);

  // attn LIF (reuse xs buffer for spikes), then final GEMM + BN -> d_out
  lif_kernel<<<LIF_BLOCKS, 256, 0, stream>>>(preact, xs);
  gemm_bn_kernel<<<ggrid, 256, 0, stream>>>(xs, wl16, l_b, l_g, l_be, out);
}
